// Block_498216206781
// MI455X (gfx1250) — compile-verified, matrix operand order checked by reading, not run on hardware
//
#include <hip/hip_runtime.h>

typedef __bf16 bf16;
typedef __attribute__((ext_vector_type(16))) __bf16 v16bf;
typedef __attribute__((ext_vector_type(8)))  float  v8f;
typedef __attribute__((ext_vector_type(4)))  int    v4i;

#define B_    2
#define S_    1024
#define D_    768
#define H_    12
#define HK_   2
#define DH_   64
#define E_    32
#define F_    3072
#define MTOK  (B_ * S_)   // 2048

// ---------------------------------------------------------------------------
// CDNA5 async global->LDS copy (ASYNCcnt path), with safe fallback.
// ---------------------------------------------------------------------------
#define HAS_ASYNC_LDS __has_builtin(__builtin_amdgcn_global_load_async_to_lds_b128)

typedef __attribute__((address_space(1))) v4i g1_v4i;
typedef __attribute__((address_space(3))) v4i l3_v4i;

__device__ __forceinline__ void async_b128(const void* g, void* l) {
#if HAS_ASYNC_LDS
  __builtin_amdgcn_global_load_async_to_lds_b128((g1_v4i*)g, (l3_v4i*)l, 0, 0);
#else
  *(uint4*)l = *(const uint4*)g;
#endif
}

__device__ __forceinline__ void async_wait() {
#if HAS_ASYNC_LDS
#if __has_builtin(__builtin_amdgcn_s_wait_asynccnt)
  __builtin_amdgcn_s_wait_asynccnt(0);
#else
  __asm__ volatile("s_wait_asynccnt 0" ::: "memory");
#endif
#endif
}

// ---------------------------------------------------------------------------
// WMMA helpers (CDNA5 wave32, 16x16x32 bf16 -> f32)
// ---------------------------------------------------------------------------
__device__ __forceinline__ v8f wmma_bf16(v16bf a, v16bf b, v8f c) {
  return __builtin_amdgcn_wmma_f32_16x16x32_bf16(false, a, false, b, (short)0, c,
                                                 false, false);
}

// Fragment from a row-major 16x32 bf16 tile (A-frag, or B-frag from an
// N-major tile).  VGPR j holds K pair k0 = 16*(j>=4) + 8*h + 2*(j&3).
__device__ __forceinline__ v16bf frag_rm_bf16(const bf16* base, int ld) {
  int lane = threadIdx.x & 31;
  int r = lane & 15, h = lane >> 4;
  v16bf f;
#pragma unroll
  for (int j = 0; j < 8; ++j) {
    int k0 = ((j & 4) << 2) + (h << 3) + ((j & 3) << 1);
    f[2 * j]     = base[r * ld + k0];
    f[2 * j + 1] = base[r * ld + k0 + 1];
  }
  return f;
}

// B-fragment from a K-major (row-major [K,N]) bf16 tile: element(k, n=r) at
// base[k*ld + r].
__device__ __forceinline__ v16bf frag_kn_bf16(const bf16* base, int ld) {
  int lane = threadIdx.x & 31;
  int r = lane & 15, h = lane >> 4;
  v16bf f;
#pragma unroll
  for (int j = 0; j < 8; ++j) {
    int k0 = ((j & 4) << 2) + (h << 3) + ((j & 3) << 1);
    f[2 * j]     = base[k0 * ld + r];
    f[2 * j + 1] = base[(k0 + 1) * ld + r];
  }
  return f;
}

__device__ __forceinline__ float gelu_tanh(float x) {
  float x3 = x * x * x;
  float t = tanhf(0.7978845608028654f * (x + 0.044715f * x3));
  return 0.5f * x * (1.f + t);
}

// ---------------------------------------------------------------------------
// RMSNorm (f32 in) -> bf16 out.  One wave per token row.
// ---------------------------------------------------------------------------
__global__ __launch_bounds__(256) void k_rmsnorm(const float* __restrict__ x,
                                                 const float* __restrict__ wt,
                                                 bf16* __restrict__ out) {
  int w = threadIdx.x >> 5, lane = threadIdx.x & 31;
  int row = blockIdx.x * 8 + w;
  if (row >= MTOK) return;
  const float* xr = x + (size_t)row * D_;
  float ss = 0.f;
  for (int c = lane; c < D_; c += 32) { float v = xr[c]; ss += v * v; }
#pragma unroll
  for (int o = 16; o; o >>= 1) ss += __shfl_xor(ss, o, 32);
  float rs = rsqrtf(ss * (1.f / D_) + 1e-6f);
  for (int c = lane; c < D_; c += 32)
    out[(size_t)row * D_ + c] = (bf16)(xr[c] * rs * wt[c]);
}

// ---------------------------------------------------------------------------
// Generic GEMM: C[M,N] = A[M,K](bf16) * W[K,N](f32 -> bf16 in staging).
// mode 0: store bf16 C;  mode 1: store f32 C;  mode 2: f32 C = res + acc.
// Block 256 threads = 8 waves, tile 64x64, each wave a 16x32 strip.
// Requires M % 64 == 0 and K % 32 == 0 (true for all call sites).
// ---------------------------------------------------------------------------
__global__ __launch_bounds__(256) void k_gemm(const bf16* __restrict__ A,
                                              const float* __restrict__ W,
                                              int M, int N, int K,
                                              float* __restrict__ Cf,
                                              bf16* __restrict__ Cb,
                                              const float* __restrict__ res,
                                              int mode) {
  __shared__ bf16 As[64][32];
  __shared__ bf16 Bt[64][32];  // [n][k]
  int tid = threadIdx.x;
  int lane = tid & 31, w = tid >> 5;
  int r = lane & 15, h = lane >> 4;
  int m0 = blockIdx.x * 64, n0 = blockIdx.y * 64;
  int wm = w & 3, wn = w >> 2;
  int rowa = tid >> 2, sega = (tid & 3) * 8;  // A tile: 256 x 16B segments
  v8f acc0 = {}, acc1 = {};
  for (int kk = 0; kk < K; kk += 32) {
    __syncthreads();
    // A tile via async-to-LDS B128
    async_b128(A + (size_t)(m0 + rowa) * K + kk + sega, &As[rowa][sega]);
    // W tile 32x64 f32, coalesced float4 loads, transposed bf16 into Bt
    for (int i = tid; i < 32 * 16; i += 256) {
      int rk = i >> 4, c4 = (i & 15) << 2;
      int gn = n0 + c4;
      const float* wp = W + (size_t)(kk + rk) * N + gn;
      float w0 = 0.f, w1 = 0.f, w2 = 0.f, w3 = 0.f;
      if (gn + 3 < N) {
        float4 t = *(const float4*)wp;
        w0 = t.x; w1 = t.y; w2 = t.z; w3 = t.w;
      } else {
        if (gn < N)     w0 = wp[0];
        if (gn + 1 < N) w1 = wp[1];
        if (gn + 2 < N) w2 = wp[2];
        if (gn + 3 < N) w3 = wp[3];
      }
      Bt[c4 + 0][rk] = (bf16)w0; Bt[c4 + 1][rk] = (bf16)w1;
      Bt[c4 + 2][rk] = (bf16)w2; Bt[c4 + 3][rk] = (bf16)w3;
    }
    async_wait();
    __syncthreads();
    v16bf af = frag_rm_bf16(&As[wm * 16][0], 32);
    v16bf b0 = frag_rm_bf16(&Bt[wn * 32][0], 32);
    v16bf b1 = frag_rm_bf16(&Bt[wn * 32 + 16][0], 32);
    acc0 = wmma_bf16(af, b0, acc0);
    acc1 = wmma_bf16(af, b1, acc1);
  }
#pragma unroll
  for (int i = 0; i < 8; ++i) {
    int row = m0 + wm * 16 + i + 8 * h;
    if (row >= M) continue;
    int c0 = n0 + wn * 32 + r, c1 = c0 + 16;
    float v0 = acc0[i], v1 = acc1[i];
    if (mode == 0) {
      if (c0 < N) Cb[(size_t)row * N + c0] = (bf16)v0;
      if (c1 < N) Cb[(size_t)row * N + c1] = (bf16)v1;
    } else if (mode == 1) {
      if (c0 < N) Cf[(size_t)row * N + c0] = v0;
      if (c1 < N) Cf[(size_t)row * N + c1] = v1;
    } else {
      if (c0 < N) Cf[(size_t)row * N + c0] = res[(size_t)row * N + c0] + v0;
      if (c1 < N) Cf[(size_t)row * N + c1] = res[(size_t)row * N + c1] + v1;
    }
  }
}

// ---------------------------------------------------------------------------
// Flash-style causal GQA attention.  Block = 128 threads (4 waves), one wave
// per 16 query rows.  grid = (B*H, S/64).
// ---------------------------------------------------------------------------
__global__ __launch_bounds__(128) void k_attn(const bf16* __restrict__ qm,
                                              const bf16* __restrict__ km,
                                              const bf16* __restrict__ vm,
                                              bf16* __restrict__ om) {
  __shared__ bf16 Pbuf[4][16][32];
  int tid = threadIdx.x, w = tid >> 5, lane = tid & 31;
  int r = lane & 15, h = lane >> 4;
  int bh = blockIdx.x;
  int b = bh / H_, hh = bh % H_, kvh = hh / (H_ / HK_);
  int q0 = blockIdx.y * 64 + w * 16;

  const bf16* qbase = qm + (size_t)(b * S_ + q0) * D_ + hh * DH_;
  v16bf qf0 = frag_rm_bf16(qbase, D_);
  v16bf qf1 = frag_rm_bf16(qbase + 32, D_);

  float mrow[8], lrow[8];
  v8f oacc[4] = {};
#pragma unroll
  for (int i = 0; i < 8; ++i) { mrow[i] = -3e38f; lrow[i] = 0.f; }

  int kvend = q0 + 16;  // max kv needed (causal, exclusive)
  for (int kv0 = 0; kv0 < kvend; kv0 += 32) {
    const bf16* kb = km + (size_t)(b * S_ + kv0) * (HK_ * DH_) + kvh * DH_;
    v8f s0 = {}, s1 = {};
    s0 = wmma_bf16(qf0, frag_rm_bf16(kb, HK_ * DH_), s0);
    s0 = wmma_bf16(qf1, frag_rm_bf16(kb + 32, HK_ * DH_), s0);
    s1 = wmma_bf16(qf0, frag_rm_bf16(kb + 16 * (HK_ * DH_), HK_ * DH_), s1);
    s1 = wmma_bf16(qf1, frag_rm_bf16(kb + 16 * (HK_ * DH_) + 32, HK_ * DH_), s1);

#pragma unroll
    for (int i = 0; i < 8; ++i) {
      int qrow = q0 + i + 8 * h;
      float v0 = (kv0 + r <= qrow) ? s0[i] * 0.125f : -3e38f;
      float v1 = (kv0 + 16 + r <= qrow) ? s1[i] * 0.125f : -3e38f;
      float mx = fmaxf(v0, v1);
#pragma unroll
      for (int off = 1; off < 16; off <<= 1) mx = fmaxf(mx, __shfl_xor(mx, off, 16));
      float mn = fmaxf(mrow[i], mx);
      float al = __expf(mrow[i] - mn);
      float p0 = __expf(v0 - mn), p1 = __expf(v1 - mn);
      float ps = p0 + p1;
#pragma unroll
      for (int off = 1; off < 16; off <<= 1) ps += __shfl_xor(ps, off, 16);
      lrow[i] = lrow[i] * al + ps;
      mrow[i] = mn;
#pragma unroll
      for (int t = 0; t < 4; ++t) oacc[t][i] *= al;
      Pbuf[w][i + 8 * h][r]      = (bf16)p0;
      Pbuf[w][i + 8 * h][16 + r] = (bf16)p1;
    }
    __asm__ volatile("s_wait_dscnt 0" ::: "memory");  // LDS transpose hand-off
    v16bf pf = frag_rm_bf16(&Pbuf[w][0][0], 32);
    const bf16* vb = vm + (size_t)(b * S_ + kv0) * (HK_ * DH_) + kvh * DH_;
#pragma unroll
    for (int t = 0; t < 4; ++t)
      oacc[t] = wmma_bf16(pf, frag_kn_bf16(vb + t * 16, HK_ * DH_), oacc[t]);
  }
#pragma unroll
  for (int t = 0; t < 4; ++t)
#pragma unroll
    for (int i = 0; i < 8; ++i) {
      int row = b * S_ + q0 + i + 8 * h;
      int col = hh * DH_ + t * 16 + r;
      om[(size_t)row * D_ + col] = (bf16)(oacc[t][i] / lrow[i]);
    }
}

// ---------------------------------------------------------------------------
// Router softmax + top-2 + expert gather + aux-loss partials.
// ---------------------------------------------------------------------------
__global__ __launch_bounds__(256) void k_topk(const float* __restrict__ logits,
                                              int* __restrict__ cnt,
                                              int* __restrict__ toklist,
                                              float* __restrict__ coef,
                                              float* __restrict__ psum) {
  int t = blockIdx.x * blockDim.x + threadIdx.x;
  if (t >= MTOK) return;
  const float* lg = logits + (size_t)t * E_;
  float mx = -3e38f;
#pragma unroll
  for (int e = 0; e < E_; ++e) mx = fmaxf(mx, lg[e]);
  float ex[E_]; float s = 0.f;
#pragma unroll
  for (int e = 0; e < E_; ++e) { ex[e] = __expf(lg[e] - mx); s += ex[e]; }
  float inv = 1.f / s;
  int i1 = -1, i2 = -1; float v1 = -1.f, v2 = -1.f;
#pragma unroll
  for (int e = 0; e < E_; ++e) {
    float p = ex[e] * inv;
    atomicAdd(&psum[e], p);
    if (p > v1) { v2 = v1; i2 = i1; v1 = p; i1 = e; }
    else if (p > v2) { v2 = p; i2 = e; }
  }
  float wn = 1.f / (v1 + v2);
  int p1 = atomicAdd(&cnt[i1], 1);
  toklist[i1 * MTOK + p1] = t; coef[i1 * MTOK + p1] = v1 * wn;
  int p2 = atomicAdd(&cnt[i2], 1);
  toklist[i2 * MTOK + p2] = t; coef[i2 * MTOK + p2] = v2 * wn;
}

__global__ void k_aux(const int* __restrict__ cnt, const float* __restrict__ psum,
                      float* __restrict__ out_aux) {
  if (threadIdx.x == 0 && blockIdx.x == 0) {
    float a = 0.f;
    for (int e = 0; e < E_; ++e)
      a += (cnt[e] * (1.f / MTOK)) * (psum[e] * (1.f / MTOK));
    *out_aux = (float)E_ * a;
  }
}

// ---------------------------------------------------------------------------
// Fused per-expert MoE: out += coef * (gelu(h2 @ W1_e) @ W2_e) for gathered
// tokens.  Block = 256 threads (8 waves) handles a 16-token tile; F chunks of
// 512 with stage-2 accumulators persistent in VGPRs.  Weight k-slabs are
// cooperatively staged into LDS (coalesced f32 float4 -> bf16, transposed).
// grid = (E, MTOK/16), empty tiles exit early (uniform per block).
// ---------------------------------------------------------------------------
__global__ __launch_bounds__(256) void k_moe(const bf16* __restrict__ h2,
                                             const float* __restrict__ W1,
                                             const float* __restrict__ W2,
                                             const int* __restrict__ cnt,
                                             const int* __restrict__ toklist,
                                             const float* __restrict__ coef,
                                             float* __restrict__ out) {
  int e = blockIdx.x, tile = blockIdx.y;
  int ntok = cnt[e];
  if (tile * 16 >= ntok) return;

  __shared__ bf16 Ah[16][D_];       // 24 KB  token activations
  __shared__ bf16 Act[16][512];     // 16 KB  gelu(h2@W1) chunk
  __shared__ bf16 Bw[768 * 32];     // 48 KB  [n][k] weight slab (shared W1/W2)
  __shared__ int   stok[16];
  __shared__ float scoef[16];

  int tid = threadIdx.x, w = tid >> 5, lane = tid & 31;
  int r = lane & 15, h = lane >> 4;

  if (tid < 16) {
    int slot = tile * 16 + tid;
    stok[tid]  = (slot < ntok) ? toklist[e * MTOK + slot] : -1;
    scoef[tid] = (slot < ntok) ? coef[e * MTOK + slot] : 0.f;
  }
  __syncthreads();
  // gather token rows (async-to-LDS for valid rows, zero-fill otherwise)
  for (int i = tid; i < 16 * (D_ / 8); i += 256) {  // 16B segments
    int rr = i / (D_ / 8), sg = (i % (D_ / 8)) * 8;
    int tok = stok[rr];
    if (tok >= 0) {
      async_b128(h2 + (size_t)tok * D_ + sg, &Ah[rr][sg]);
    } else {
      uint4 z = {0u, 0u, 0u, 0u};
      *(uint4*)&Ah[rr][sg] = z;
    }
  }
  async_wait();
  __syncthreads();

  const float* W1e = W1 + (size_t)e * D_ * F_;
  const float* W2e = W2 + (size_t)e * F_ * D_;

  v8f oaccs[6] = {};
  for (int f0 = 0; f0 < F_; f0 += 512) {
    // ---- stage 1: Act[16][512] = gelu(Ah @ W1_e[:, f0:f0+512]) ----
    int cbase = w * 64;  // this wave's 64 columns within the chunk
    v8f aarr[4] = {};
    for (int kk = 0; kk < D_; kk += 32) {
      __syncthreads();  // previous slab fully consumed
      for (int i = tid; i < 32 * 128; i += 256) {  // 32k x 512n f32 slab
        int rk = i >> 7, c4 = (i & 127) << 2;
        const float* wp = W1e + (size_t)(kk + rk) * F_ + f0 + c4;
        float4 t = *(const float4*)wp;
        Bw[(c4 + 0) * 32 + rk] = (bf16)t.x;
        Bw[(c4 + 1) * 32 + rk] = (bf16)t.y;
        Bw[(c4 + 2) * 32 + rk] = (bf16)t.z;
        Bw[(c4 + 3) * 32 + rk] = (bf16)t.w;
      }
      __syncthreads();
      v16bf af = frag_rm_bf16(&Ah[0][kk], D_);
#pragma unroll
      for (int t = 0; t < 4; ++t)
        aarr[t] = wmma_bf16(af, frag_rm_bf16(&Bw[(cbase + t * 16) * 32], 32),
                            aarr[t]);
    }
    __syncthreads();
#pragma unroll
    for (int t = 0; t < 4; ++t)
#pragma unroll
      for (int i = 0; i < 8; ++i)
        Act[i + 8 * h][cbase + t * 16 + r] = (bf16)gelu_tanh(aarr[t][i]);
    __syncthreads();
    // ---- stage 2 partial: out += Act @ W2_e[f0:f0+512, :] ----
    int nb = w * 96;  // this wave's 96 output columns
    for (int kk = 0; kk < 512; kk += 32) {
      __syncthreads();  // previous slab fully consumed
      for (int i = tid; i < 32 * 192; i += 256) {  // 32k x 768n f32 slab
        int rk = i / 192, c4 = (i % 192) << 2;
        const float* wp = W2e + (size_t)(f0 + kk + rk) * D_ + c4;
        float4 t = *(const float4*)wp;
        Bw[(c4 + 0) * 32 + rk] = (bf16)t.x;
        Bw[(c4 + 1) * 32 + rk] = (bf16)t.y;
        Bw[(c4 + 2) * 32 + rk] = (bf16)t.z;
        Bw[(c4 + 3) * 32 + rk] = (bf16)t.w;
      }
      __syncthreads();
      v16bf pf = frag_rm_bf16(&Act[0][kk], 512);
#pragma unroll
      for (int t = 0; t < 6; ++t)
        oaccs[t] = wmma_bf16(pf, frag_rm_bf16(&Bw[(nb + t * 16) * 32], 32),
                             oaccs[t]);
    }
  }

#pragma unroll
  for (int i = 0; i < 8; ++i) {
    int row = i + 8 * h;
    int tok = stok[row];
    if (tok < 0) continue;
    float c = scoef[row];
#pragma unroll
    for (int t = 0; t < 6; ++t) {
      int col = w * 96 + t * 16 + r;
      atomicAdd(&out[(size_t)tok * D_ + col], c * oaccs[t][i]);
    }
  }
}

// ---------------------------------------------------------------------------
// Host-side orchestration
// ---------------------------------------------------------------------------
extern "C" void kernel_launch(void* const* d_in, const int* in_sizes, int n_in,
                              void* d_out, int out_size, void* d_ws, size_t ws_size,
                              hipStream_t stream) {
  (void)in_sizes; (void)n_in; (void)out_size; (void)ws_size;
  const float* x        = (const float*)d_in[0];
  const float* norm1_w  = (const float*)d_in[1];
  const float* norm2_w  = (const float*)d_in[2];
  const float* Wq       = (const float*)d_in[3];
  const float* Wk       = (const float*)d_in[4];
  const float* Wv       = (const float*)d_in[5];
  const float* Wo       = (const float*)d_in[6];
  const float* router_w = (const float*)d_in[7];
  const float* W1       = (const float*)d_in[8];
  const float* W2       = (const float*)d_in[9];

  char* ws = (char*)d_ws;
  size_t off = 0;
  auto carve = [&](size_t bytes) -> char* {
    char* p = ws + off;
    off = (off + bytes + 255) & ~(size_t)255;
    return p;
  };
  bf16*  h_bf   = (bf16*)carve((size_t)MTOK * D_ * 2);
  bf16*  q_bf   = (bf16*)carve((size_t)MTOK * D_ * 2);
  bf16*  k_bf   = (bf16*)carve((size_t)MTOK * HK_ * DH_ * 2);
  bf16*  v_bf   = (bf16*)carve((size_t)MTOK * HK_ * DH_ * 2);
  bf16*  o_bf   = (bf16*)carve((size_t)MTOK * D_ * 2);
  float* x1     = (float*)carve((size_t)MTOK * D_ * 4);
  bf16*  h2_bf  = (bf16*)carve((size_t)MTOK * D_ * 2);
  float* logits = (float*)carve((size_t)MTOK * E_ * 4);
  int*   cnt    = (int*)carve(E_ * 4);
  float* psum   = (float*)carve(E_ * 4);
  int*   toklst = (int*)carve((size_t)E_ * MTOK * 4);
  float* coefl  = (float*)carve((size_t)E_ * MTOK * 4);

  float* outp = (float*)d_out;

  // 1) h = rms(x, norm1_w) -> bf16
  k_rmsnorm<<<MTOK / 8, 256, 0, stream>>>(x, norm1_w, h_bf);
  // 2) q/k/v projections
  k_gemm<<<dim3(MTOK / 64, D_ / 64), 256, 0, stream>>>(h_bf, Wq, MTOK, D_, D_,
                                                       nullptr, q_bf, nullptr, 0);
  k_gemm<<<dim3(MTOK / 64, (HK_ * DH_) / 64), 256, 0, stream>>>(
      h_bf, Wk, MTOK, HK_ * DH_, D_, nullptr, k_bf, nullptr, 0);
  k_gemm<<<dim3(MTOK / 64, (HK_ * DH_) / 64), 256, 0, stream>>>(
      h_bf, Wv, MTOK, HK_ * DH_, D_, nullptr, v_bf, nullptr, 0);
  // 3) causal GQA flash attention
  k_attn<<<dim3(B_ * H_, S_ / 64), 128, 0, stream>>>(q_bf, k_bf, v_bf, o_bf);
  // 4) x1 = x + o @ Wo
  k_gemm<<<dim3(MTOK / 64, D_ / 64), 256, 0, stream>>>(o_bf, Wo, MTOK, D_, D_,
                                                       x1, nullptr, x, 2);
  // 5) seed output with x1 (MoE accumulates on top)
  (void)hipMemcpyAsync(outp, x1, (size_t)MTOK * D_ * 4, hipMemcpyDeviceToDevice,
                       stream);
  // 6) h2 = rms(x1, norm2_w) -> bf16
  k_rmsnorm<<<MTOK / 8, 256, 0, stream>>>(x1, norm2_w, h2_bf);
  // 7) router logits
  k_gemm<<<dim3(MTOK / 64, 1), 256, 0, stream>>>(h2_bf, router_w, MTOK, E_, D_,
                                                 logits, nullptr, nullptr, 1);
  // 8) top-2 routing + gather lists
  (void)hipMemsetAsync(cnt, 0, E_ * 4, stream);
  (void)hipMemsetAsync(psum, 0, E_ * 4, stream);
  k_topk<<<MTOK / 256, 256, 0, stream>>>(logits, cnt, toklst, coefl, psum);
  // 9) sparse MoE, accumulate into d_out
  k_moe<<<dim3(E_, MTOK / 16), 256, 0, stream>>>(h2_bf, W1, W2, cnt, toklst,
                                                 coefl, outp);
  // 10) aux loss scalar
  k_aux<<<1, 32, 0, stream>>>(cnt, psum, outp + (size_t)MTOK * D_);
}
